// Net_41807211660019
// MI455X (gfx1250) — compile-verified, matrix operand order checked by reading, not run on hardware
//
#include <hip/hip_runtime.h>

typedef unsigned short u16;
typedef unsigned int u32;

typedef __attribute__((ext_vector_type(16))) __bf16 v16bf;
typedef __attribute__((ext_vector_type(8)))  float  v8f;

#define NN    16384
#define EE    524288
#define DD    512
#define DFFc  1024
#define ZIc   64
#define KK    4
#define LL    4
#define GG    32
#define NPG   512   // nodes per graph

#define USE_ASYNC 1

static __device__ __forceinline__ u16 f2bf(float f) {
  u32 u = __float_as_uint(f);
  u32 r = u + 0x7FFFu + ((u >> 16) & 1u);
  return (u16)(r >> 16);
}
static __device__ __forceinline__ float bf2f(u16 b) {
  return __uint_as_float(((u32)b) << 16);
}

#if USE_ASYNC
// CDNA5 async global->LDS copy (16B per lane), tracked by ASYNCcnt.
static __device__ __forceinline__ void async_copy_b128(u32 lds_off, const void* gptr) {
  asm volatile("global_load_async_to_lds_b128 %0, %1, off"
               :: "v"(lds_off), "v"(gptr) : "memory");
}
static __device__ __forceinline__ void wait_async0() {
#if __has_builtin(__builtin_amdgcn_s_wait_asynccnt)
  __builtin_amdgcn_s_wait_asynccnt(0);
#else
  asm volatile("s_wait_asynccnt 0x0" ::: "memory");
#endif
}
static __device__ __forceinline__ void wait_async_le2() {
#if __has_builtin(__builtin_amdgcn_s_wait_asynccnt)
  __builtin_amdgcn_s_wait_asynccnt(2);
#else
  asm volatile("s_wait_asynccnt 0x2" ::: "memory");
#endif
}
#endif

// ---------------- elementwise utility kernels (4 elems/thread) ----------------
__global__ __launch_bounds__(256) void zero4(float* __restrict__ p) {
  const int i = blockIdx.x * 256 + threadIdx.x;
  ((float4*)p)[i] = make_float4(0.f, 0.f, 0.f, 0.f);
}
__global__ __launch_bounds__(256) void conv4_f2bf(const float* __restrict__ s,
                                                  u16* __restrict__ d) {
  const int i = blockIdx.x * 256 + threadIdx.x;
  const float4 v = ((const float4*)s)[i];
  uint2 o;
  o.x = (u32)f2bf(v.x) | ((u32)f2bf(v.y) << 16);
  o.y = (u32)f2bf(v.z) | ((u32)f2bf(v.w) << 16);
  ((uint2*)d)[i] = o;
}
__global__ __launch_bounds__(256) void add_conv4(const float* __restrict__ a,
                                                 const float* __restrict__ b,
                                                 u16* __restrict__ d) {
  const int i = blockIdx.x * 256 + threadIdx.x;
  const float4 va = ((const float4*)a)[i];
  const float4 vb = ((const float4*)b)[i];
  uint2 o;
  o.x = (u32)f2bf(va.x + vb.x) | ((u32)f2bf(va.y + vb.y) << 16);
  o.y = (u32)f2bf(va.z + vb.z) | ((u32)f2bf(va.w + vb.w) << 16);
  ((uint2*)d)[i] = o;
}
// transpose-convert: dst[n*K+k] = bf16(src[k*Nc+n]); K = 1<<kshift
__global__ __launch_bounds__(256) void conv_t_f2bf(const float* __restrict__ s,
                                                   u16* __restrict__ d,
                                                   int Nc, int kshift) {
  const int o = blockIdx.x * 256 + threadIdx.x;
  const int K = 1 << kshift;
  const int n = o >> kshift;
  const int k = o & (K - 1);
  d[o] = f2bf(s[(size_t)k * Nc + n]);
}

// ---------------- node embedding: h = x @ emb_w + emb_b ----------------
__global__ __launch_bounds__(256) void embed_k(const float* __restrict__ x,
                                               const float* __restrict__ w,
                                               const float* __restrict__ b,
                                               float* __restrict__ h) {
  __shared__ float xr[9];
  const int i = blockIdx.x, t = threadIdx.x;
  if (t < 9) xr[t] = x[i * 9 + t];
  __syncthreads();
  for (int d = t; d < DD; d += 256) {
    float s = b[d];
#pragma unroll
    for (int j = 0; j < 9; ++j) s += xr[j] * w[j * DD + d];
    h[(size_t)i * DD + d] = s;
  }
}

// ---------------- edge scatter: agg[dst] += h[src] ----------------
__global__ __launch_bounds__(256) void scatter_agg(const float* __restrict__ h,
                                                   const long long* __restrict__ ei,
                                                   float* __restrict__ agg) {
  const int e = blockIdx.x * 2 + (threadIdx.x >> 7);  // 128 threads per edge
  const int t = threadIdx.x & 127;
  const long long s = ei[e];
  const long long d = ei[EE + e];
  const float* hs = h + (size_t)s * DD;
  float* ad = agg + (size_t)d * DD;
#pragma unroll
  for (int i = 0; i < 4; ++i)
    atomicAdd(&ad[t + i * 128], hs[t + i * 128]);
}

// ---------------- WMMA GEMM: C[M,Nc] = A[M,K](bf16) @ Bt[Nc,K](bf16, pre-transposed) --
// block = 256 threads = 8 waves; block tile 64x64 (4x2 waves, each 16x32 via 2 accs).
// Double-buffered async global->LDS staging. Requires M%64==0, Nc%64==0, K%32==0.
template <bool RELU, bool OUTF32>
__global__ __launch_bounds__(256) void gemm_bf16(const u16* __restrict__ A,
                                                 const u16* __restrict__ Bt,
                                                 const float* __restrict__ bias,
                                                 float* __restrict__ Cf,
                                                 u16* __restrict__ Cbf,
                                                 int K, int Nc) {
  __shared__ __align__(16) u16 As[2][64 * 32];  // [buf][row][k]
  __shared__ __align__(16) u16 Bs[2][64 * 32];  // [buf][col][k]
  const int t = threadIdx.x;
  const int wid = t >> 5, lane = t & 31;
  const int wm = wid & 3, wn = wid >> 2;          // wm: 0..3 rows, wn: 0..1 cols
  const int half = lane >> 4, l15 = lane & 15;
  const int row0 = blockIdx.y * 64;
  const int col0 = blockIdx.x * 64;

  const int sr = t >> 2, sc = (t & 3) * 8;        // staging: 1 uint4 per thread per tile
  const u16* ga = A  + (size_t)(row0 + sr) * K + sc;
  const u16* gb = Bt + (size_t)(col0 + sr) * K + sc;
  const int soff = sr * 32 + sc;
  const int T = K >> 5;

#if USE_ASYNC
  async_copy_b128((u32)(size_t)(&As[0][soff]), ga);
  async_copy_b128((u32)(size_t)(&Bs[0][soff]), gb);
#endif

  v8f acc0 = {}, acc1 = {};
  for (int i = 0; i < T; ++i) {
    const int cur = i & 1;
#if USE_ASYNC
    if (i + 1 < T) {  // prefetch next tile into alternate buffer
      async_copy_b128((u32)(size_t)(&As[cur ^ 1][soff]), ga + (i + 1) * 32);
      async_copy_b128((u32)(size_t)(&Bs[cur ^ 1][soff]), gb + (i + 1) * 32);
      wait_async_le2();  // in-order completion => tile i resident
    } else {
      wait_async0();
    }
#else
    *(uint4*)(&As[cur][soff]) = *(const uint4*)(ga + i * 32);
    *(uint4*)(&Bs[cur][soff]) = *(const uint4*)(gb + i * 32);
#endif
    __syncthreads();

    union { uint4 q[2]; v16bf v; } af, bf0, bf1;
    const u16* ap = &As[cur][(wm * 16 + l15) * 32 + half * 8];
    af.q[0] = *(const uint4*)(ap);
    af.q[1] = *(const uint4*)(ap + 16);
    const u16* bp0 = &Bs[cur][(wn * 32 + l15) * 32 + half * 8];
    bf0.q[0] = *(const uint4*)(bp0);
    bf0.q[1] = *(const uint4*)(bp0 + 16);
    const u16* bp1 = bp0 + 16 * 32;
    bf1.q[0] = *(const uint4*)(bp1);
    bf1.q[1] = *(const uint4*)(bp1 + 16);

    acc0 = __builtin_amdgcn_wmma_f32_16x16x32_bf16(false, af.v, false, bf0.v,
                                                   (short)0, acc0, false, false);
    acc1 = __builtin_amdgcn_wmma_f32_16x16x32_bf16(false, af.v, false, bf1.v,
                                                   (short)0, acc1, false, false);
    __syncthreads();  // all reads of buf `cur` done before it is refilled
  }

  const int gcol0 = col0 + wn * 32 + l15;
  const int gcol1 = gcol0 + 16;
  const float bv0 = bias[gcol0];
  const float bv1 = bias[gcol1];
#pragma unroll
  for (int j = 0; j < 8; ++j) {
    const int grow = row0 + wm * 16 + j + 8 * half;
    float v0 = acc0[j] + bv0;
    float v1 = acc1[j] + bv1;
    if (RELU) { v0 = fmaxf(v0, 0.0f); v1 = fmaxf(v1, 0.0f); }
    if (OUTF32) {
      Cf[(size_t)grow * Nc + gcol0] = v0;
      Cf[(size_t)grow * Nc + gcol1] = v1;
    } else {
      Cbf[(size_t)grow * Nc + gcol0] = f2bf(v0);
      Cbf[(size_t)grow * Nc + gcol1] = f2bf(v1);
    }
  }
}

// ---------------- BatchNorm stats (column sum / sumsq over N rows) ----------------
__global__ __launch_bounds__(256) void bn_stats(const float* __restrict__ m,
                                                float* __restrict__ stats) {
  __shared__ float ssum[DD];
  __shared__ float ssq[DD];
  const int t = threadIdx.x;
  for (int s = t; s < DD; s += 256) { ssum[s] = 0.f; ssq[s] = 0.f; }
  __syncthreads();
  const size_t base = (size_t)blockIdx.x * 64 * DD;  // 64 rows per block
  for (int i = 0; i < 128; ++i) {
    const size_t idx = base + t + (size_t)i * 256;
    const float v = m[idx];
    const int d = (int)(idx & (DD - 1));
    atomicAdd(&ssum[d], v);
    atomicAdd(&ssq[d], v * v);
  }
  __syncthreads();
  for (int s = t; s < DD; s += 256) {
    atomicAdd(&stats[s], ssum[s]);
    atomicAdd(&stats[DD + s], ssq[s]);
  }
}

// ---------------- BN apply + ReLU + residual: h += relu(bn(m)) ----------------
__global__ __launch_bounds__(256) void bn_apply(const float* __restrict__ m,
                                                const float* __restrict__ stats,
                                                const float* __restrict__ g,
                                                const float* __restrict__ b,
                                                float* __restrict__ h) {
  const size_t idx = (size_t)blockIdx.x * 256 + threadIdx.x;
  const int d = (int)(idx & (DD - 1));
  const float invN = 1.0f / (float)NN;
  const float mu = stats[d] * invN;
  const float var = stats[DD + d] * invN - mu * mu;
  const float sc = rsqrtf(var + 1e-5f) * g[d];
  const float v = (m[idx] - mu) * sc + b[d];
  h[idx] += fmaxf(v, 0.0f);
}

// ---------------- row norms of Hk (bf16) ----------------
__global__ __launch_bounds__(256) void row_norms(const u16* __restrict__ hk,
                                                 float* __restrict__ r) {
  const int row = blockIdx.x * 256 + threadIdx.x;  // K*N rows
  const u16* p = hk + (size_t)row * ZIc;
  float s = 0.f;
#pragma unroll
  for (int z = 0; z < ZIc; ++z) { const float f = bf2f(p[z]); s += f * f; }
  r[row] = s;
}

// ---------------- pairwise UMAP q via WMMA Gram ----------------
// grid: (n/32, n/64, G*K); block 256 = 8 waves -> 64x32 output tile per block
__global__ __launch_bounds__(256) void pairwise_q(const u16* __restrict__ hk,
                                                  const float* __restrict__ r,
                                                  float* __restrict__ out) {
  const int t = threadIdx.x;
  const int wid = t >> 5, lane = t & 31;
  const int wm = wid & 3, wn = wid >> 2;
  const int half = lane >> 4, l15 = lane & 15;
  const int gk = blockIdx.z;
  const int k = gk & 3, g = gk >> 2;
  const size_t base = ((size_t)k * NN + (size_t)g * NPG) * ZIc;
  const int row0 = blockIdx.y * 64;
  const int col0 = blockIdx.x * 32;
  const int arow = row0 + wm * 16 + l15;
  const int bcol = col0 + wn * 16 + l15;

  v8f acc = {};
#pragma unroll
  for (int k0 = 0; k0 < ZIc; k0 += 32) {
    union { uint4 q[2]; v16bf v; } af, bfr;
    const u16* ap = hk + base + (size_t)arow * ZIc + k0 + half * 8;
    const u16* bp = hk + base + (size_t)bcol * ZIc + k0 + half * 8;
    af.q[0] = *(const uint4*)(ap);
    af.q[1] = *(const uint4*)(ap + 16);
    bfr.q[0] = *(const uint4*)(bp);
    bfr.q[1] = *(const uint4*)(bp + 16);
    acc = __builtin_amdgcn_wmma_f32_16x16x32_bf16(false, af.v, false, bfr.v,
                                                  (short)0, acc, false, false);
  }

  const size_t rbase = (size_t)k * NN + (size_t)g * NPG;
  const float rj = r[rbase + bcol];
  const size_t obase = (((size_t)g * KK + k) * NPG) * NPG;
#pragma unroll
  for (int j = 0; j < 8; ++j) {
    const int irow = row0 + wm * 16 + j + 8 * half;
    const float ri = r[rbase + irow];
    float d2 = ri + rj - 2.0f * acc[j];
    d2 = fmaxf(d2, 1e-12f);
    const float q = 1.0f / (1.0f + 1.577f * __expf(0.8951f * __logf(d2)));
    out[obase + (size_t)irow * NPG + bcol] = q;
  }
}

// ---------------- host side ----------------
static inline void launch_gemm(const u16* A, const u16* Bt, const float* bias,
                               float* Cf, u16* Cbf, int M, int K, int Nc,
                               int relu, hipStream_t s) {
  dim3 grid(Nc / 64, M / 64);
  if (Cf) {
    if (relu) gemm_bf16<true, true><<<grid, 256, 0, s>>>(A, Bt, bias, Cf, Cbf, K, Nc);
    else      gemm_bf16<false, true><<<grid, 256, 0, s>>>(A, Bt, bias, Cf, Cbf, K, Nc);
  } else {
    if (relu) gemm_bf16<true, false><<<grid, 256, 0, s>>>(A, Bt, bias, Cf, Cbf, K, Nc);
    else      gemm_bf16<false, false><<<grid, 256, 0, s>>>(A, Bt, bias, Cf, Cbf, K, Nc);
  }
}

extern "C" void kernel_launch(void* const* d_in, const int* in_sizes, int n_in,
                              void* d_out, int out_size, void* d_ws, size_t ws_size,
                              hipStream_t stream) {
  (void)in_sizes; (void)n_in; (void)out_size; (void)ws_size;
  const float* x       = (const float*)d_in[0];
  const float* emb_w   = (const float*)d_in[2];
  const float* emb_b   = (const float*)d_in[3];
  const float* gin_w1  = (const float*)d_in[4];
  const float* gin_b1  = (const float*)d_in[5];
  const float* gin_w2  = (const float*)d_in[6];
  const float* gin_b2  = (const float*)d_in[7];
  const float* bn_g    = (const float*)d_in[8];
  const float* bn_b    = (const float*)d_in[9];
  const float* mlp_w1  = (const float*)d_in[10];
  const float* mlp_b1  = (const float*)d_in[11];
  const float* mlp_w2  = (const float*)d_in[12];
  const float* mlp_b2  = (const float*)d_in[13];
  const float* mlp_w3  = (const float*)d_in[14];
  const float* mlp_b3  = (const float*)d_in[15];
  const float* head_w1 = (const float*)d_in[16];
  const float* head_b1 = (const float*)d_in[17];
  const float* head_w2 = (const float*)d_in[18];
  const float* head_b2 = (const float*)d_in[19];
  const long long* ei  = (const long long*)d_in[20];
  float* out = (float*)d_out;

  const size_t MB = 1ull << 20;
  char* ws = (char*)d_ws;
  float* h     = (float*)(ws + 0 * MB);            // N*D f32   (32MB)
  float* agg   = (float*)(ws + 32 * MB);           // N*D f32   (32MB)
  float* mbuf  = (float*)(ws + 64 * MB);           // N*D f32   (32MB)
  float* stats = (float*)(ws + 96 * MB);           // 1024 f32
  float* rbuf  = (float*)(ws + 96 * MB + 16384);   // 65536 f32
  u16* bfa = (u16*)(ws + 97 * MB);                 // 16M bf16  (32MB)
  u16* bfb = (u16*)(ws + 129 * MB);                // 16M bf16  (32MB)
  u16* wbf = (u16*)(ws + 161 * MB);                // 4M bf16   (8MB)

  // weight bf16 (transposed [Nc][K]) offsets, elements
  const size_t GW1 = 0, GW2 = 1048576, MW1 = 2097152, MW2 = 2621440,
               MW3 = 3670016, HW1 = 3735552, HW2 = 3751936;

  // convert + transpose all weights once
  for (int l = 0; l < LL; ++l) {
    conv_t_f2bf<<<(DD * DD) / 256, 256, 0, stream>>>(
        gin_w1 + (size_t)l * DD * DD, wbf + GW1 + (size_t)l * DD * DD, DD, 9);
    conv_t_f2bf<<<(DD * DD) / 256, 256, 0, stream>>>(
        gin_w2 + (size_t)l * DD * DD, wbf + GW2 + (size_t)l * DD * DD, DD, 9);
  }
  conv_t_f2bf<<<(DD * DFFc) / 256, 256, 0, stream>>>(mlp_w1, wbf + MW1, DFFc, 9);
  conv_t_f2bf<<<(DFFc * DFFc) / 256, 256, 0, stream>>>(mlp_w2, wbf + MW2, DFFc, 10);
  conv_t_f2bf<<<(DFFc * ZIc) / 256, 256, 0, stream>>>(mlp_w3, wbf + MW3, ZIc, 10);
  for (int k = 0; k < KK; ++k) {
    conv_t_f2bf<<<(ZIc * ZIc) / 256, 256, 0, stream>>>(
        head_w1 + (size_t)k * ZIc * ZIc, wbf + HW1 + (size_t)k * ZIc * ZIc, ZIc, 6);
    conv_t_f2bf<<<(ZIc * ZIc) / 256, 256, 0, stream>>>(
        head_w2 + (size_t)k * ZIc * ZIc, wbf + HW2 + (size_t)k * ZIc * ZIc, ZIc, 6);
  }

  // embed
  embed_k<<<NN, 256, 0, stream>>>(x, emb_w, emb_b, h);

  const int ND = NN * DD;  // 8M
  // GIN layers
  for (int l = 0; l < LL; ++l) {
    zero4<<<ND / 1024, 256, 0, stream>>>(agg);
    scatter_agg<<<EE / 2, 256, 0, stream>>>(h, ei, agg);
    add_conv4<<<ND / 1024, 256, 0, stream>>>(h, agg, bfa);
    launch_gemm(bfa, wbf + GW1 + (size_t)l * DD * DD, gin_b1 + l * DD,
                nullptr, bfb, NN, DD, DD, 1, stream);
    launch_gemm(bfb, wbf + GW2 + (size_t)l * DD * DD, gin_b2 + l * DD,
                mbuf, nullptr, NN, DD, DD, 0, stream);
    zero4<<<1, 256, 0, stream>>>(stats);  // 1024 floats
    bn_stats<<<NN / 64, 256, 0, stream>>>(mbuf, stats);
    bn_apply<<<ND / 256, 256, 0, stream>>>(mbuf, stats, bn_g + l * DD, bn_b + l * DD, h);
  }

  // MLP
  conv4_f2bf<<<ND / 1024, 256, 0, stream>>>(h, bfa);
  launch_gemm(bfa, wbf + MW1, mlp_b1, nullptr, bfb, NN, DD, DFFc, 1, stream);
  launch_gemm(bfb, wbf + MW2, mlp_b2, nullptr, bfa, NN, DFFc, DFFc, 1, stream);
  launch_gemm(bfa, wbf + MW3, mlp_b3, nullptr, bfb, NN, DFFc, ZIc, 0, stream);  // z -> bfb[0..1M)

  // heads
  u16* hk2 = bfb + 4 * 1048576;  // bfb[4M..8M)
  for (int k = 0; k < KK; ++k) {
    launch_gemm(bfb, wbf + HW1 + (size_t)k * ZIc * ZIc, head_b1 + k * ZIc,
                nullptr, bfa + (size_t)k * NN * ZIc, NN, ZIc, ZIc, 1, stream);
    launch_gemm(bfa + (size_t)k * NN * ZIc, wbf + HW2 + (size_t)k * ZIc * ZIc,
                head_b2 + k * ZIc, nullptr, hk2 + (size_t)k * NN * ZIc,
                NN, ZIc, ZIc, 0, stream);
  }

  // pairwise UMAP q
  row_norms<<<(KK * NN) / 256, 256, 0, stream>>>(hk2, rbuf);
  pairwise_q<<<dim3(NPG / 32, NPG / 64, GG * KK), 256, 0, stream>>>(hk2, rbuf, out);
}
